// MultiHeadAttention_79302276153668
// MI455X (gfx1250) — compile-verified, matrix operand order checked by reading, not run on hardware
//
#include <hip/hip_runtime.h>
#include <stdint.h>

typedef _Float16 half_t;
typedef __attribute__((ext_vector_type(4)))  _Float16 v4h;
typedef __attribute__((ext_vector_type(8)))  _Float16 v8h;
typedef __attribute__((ext_vector_type(16))) _Float16 v16h;
typedef __attribute__((ext_vector_type(8)))  float    v8f;
typedef __attribute__((ext_vector_type(4)))  float    v4f;

constexpr int Bc = 4, Tc = 2048, Dc = 1024, Hc = 16, HDc = 64;
constexpr int BT = Bc * Tc;          // 8192 rows
constexpr int LDSS = 72;             // padded LDS row stride in halves (144B, 16B-aligned)

// ---- CDNA5 primitives -------------------------------------------------------

static __device__ __forceinline__ v8f wmma_f16(v16h a, v16h b, v8f c) {
  // D = A(16x32 f16) x B(32x16 f16) + C(16x16 f32)
  return __builtin_amdgcn_wmma_f32_16x16x32_f16(false, a, false, b, (short)0, c,
                                                false, false);
}

// Async global->LDS 16-byte copy (ASYNCcnt-tracked, no VGPR round trip).
// LDS byte address = low 32 bits of the generic pointer (flat aperture rule).
static __device__ __forceinline__ void async_copy_b128(const half_t* g, half_t* l) {
  uint32_t lds = (uint32_t)(uintptr_t)l;
  uint64_t ga  = (uint64_t)(uintptr_t)g;
  asm volatile("global_load_async_to_lds_b128 %0, %1, off"
               :: "v"(lds), "v"(ga) : "memory");
}
static __device__ __forceinline__ void wait_async() {
  asm volatile("s_wait_asynccnt 0" ::: "memory");
}

// Load one 16x32 f16 A/B fragment (ISA 7.12.2 layout).
// lane L<16 -> row L, K {0..7}(v0-3) {16..23}(v4-7); lane>=16 -> K {8..15},{24..31}.
// B fragments: same packing with "row" = column index (base must be [n][k]).
static __device__ __forceinline__ v16h ldfrag(const half_t* base, int stride) {
  const int lane = threadIdx.x & 31;
  const int row  = lane & 15;
  const int kb   = (lane >> 4) << 3;        // 0 or 8
  const half_t* p = base + row * stride + kb;
  v8h lo = *(const v8h*)(p);
  v8h hi = *(const v8h*)(p + 16);
  v16h r;
#pragma unroll
  for (int i = 0; i < 8; ++i) { r[i] = lo[i]; r[i + 8] = hi[i]; }
  return r;
}

// ---- Kernel 0: fp32 -> fp16 bulk convert ------------------------------------
__global__ __launch_bounds__(256) void cvt_f16_kernel(
    const float* __restrict__ src, half_t* __restrict__ dst) {
  size_t i = (size_t)blockIdx.x * 256 + threadIdx.x;   // one v4 per thread
  v4f f = *(const v4f*)(src + i * 4);
  v4h h;
#pragma unroll
  for (int j = 0; j < 4; ++j) h[j] = (half_t)f[j];
  *(v4h*)(dst + i * 4) = h;
}

// ---- async stage of a 128x64 f16 tile into LDS (1024 b128 chunks) -----------
static __device__ __forceinline__ void stage_tile(
    const half_t* gsrc, size_t gstride, half_t* ldst, int tid) {
#pragma unroll
  for (int i = 0; i < 4; ++i) {
    int c = tid + i * 256;
    int row = c >> 3;
    int col = (c & 7) << 3;
    async_copy_b128(gsrc + (size_t)row * gstride + col, ldst + row * LDSS + col);
  }
}

// ---- Kernel 1: fused QKV projection  (C = x @ W^T) --------------------------
// grid = (BT/128, D/128, 3); block = 256 (8 waves). 128x128 tile, BK=64,
// double-buffered async staging. Q pre-scaled by 1/sqrt(HD); V stored
// transposed (B,H,HD,T) so P@V B-fragments are contiguous later.
__global__ __launch_bounds__(256) void qkv_proj_kernel(
    const half_t* __restrict__ xh, const half_t* __restrict__ Wqh,
    const half_t* __restrict__ Wkh, const half_t* __restrict__ Wvh,
    half_t* __restrict__ Qh, half_t* __restrict__ Kh, half_t* __restrict__ Vt) {
  __shared__ half_t As[2][128 * LDSS];
  __shared__ half_t Bs[2][128 * LDSS];

  const int tid  = threadIdx.x;
  const int lane = tid & 31;
  const int m0   = blockIdx.x * 128;
  const int n0   = blockIdx.y * 128;
  const int wsel = blockIdx.z;
  const half_t* W = (wsel == 0) ? Wqh : (wsel == 1) ? Wkh : Wvh;

  const int waveId = tid >> 5;
  const int wm = (waveId & 3) * 32;   // 4 waves in M
  const int wn = (waveId >> 2) * 64;  // 2 waves in N

  v8f acc[2][4] = {};

  stage_tile(xh + (size_t)m0 * Dc, Dc, As[0], tid);
  stage_tile(W  + (size_t)n0 * Dc, Dc, Bs[0], tid);
  wait_async();
  __syncthreads();

  for (int k0 = 0; k0 < Dc; k0 += 64) {
    const int buf = (k0 >> 6) & 1;
    if (k0 + 64 < Dc) {   // prefetch next tile into other buffer
      stage_tile(xh + (size_t)m0 * Dc + k0 + 64, Dc, As[buf ^ 1], tid);
      stage_tile(W  + (size_t)n0 * Dc + k0 + 64, Dc, Bs[buf ^ 1], tid);
    }
#pragma unroll
    for (int kc = 0; kc < 64; kc += 32) {
      v16h a0 = ldfrag(&As[buf][(wm + 0)  * LDSS + kc], LDSS);
      v16h a1 = ldfrag(&As[buf][(wm + 16) * LDSS + kc], LDSS);
      v16h b[4];
#pragma unroll
      for (int nt = 0; nt < 4; ++nt)
        b[nt] = ldfrag(&Bs[buf][(wn + nt * 16) * LDSS + kc], LDSS);
#pragma unroll
      for (int nt = 0; nt < 4; ++nt) {
        acc[0][nt] = wmma_f16(a0, b[nt], acc[0][nt]);
        acc[1][nt] = wmma_f16(a1, b[nt], acc[1][nt]);
      }
    }
    wait_async();
    __syncthreads();
  }

  // epilogue: scatter into head layouts
  const int nloc  = lane & 15;
  const int mhalf = (lane >> 4) << 3;
#pragma unroll
  for (int mt = 0; mt < 2; ++mt)
#pragma unroll
    for (int nt = 0; nt < 4; ++nt)
#pragma unroll
      for (int r = 0; r < 8; ++r) {
        int m = m0 + wm + mt * 16 + mhalf + r;
        int n = n0 + wn + nt * 16 + nloc;
        float v = acc[mt][nt][r];
        int b = m >> 11, t = m & (Tc - 1);
        int h = n >> 6,  hd = n & 63;
        if (wsel == 0)
          Qh[(((size_t)b * Hc + h) * Tc + t) * HDc + hd] = (half_t)(v * 0.125f);
        else if (wsel == 1)
          Kh[(((size_t)b * Hc + h) * Tc + t) * HDc + hd] = (half_t)v;
        else
          Vt[(((size_t)b * Hc + h) * HDc + hd) * Tc + t] = (half_t)v;
      }
}

// ---- Kernel 2: causal flash attention --------------------------------------
// grid = (B*H, T/128); block = 256 (8 waves), wave w owns q rows [q0+16w, +16).
// K/V 64-wide tiles double-buffered through async global->LDS copies.
__global__ __launch_bounds__(256) void flash_attn_kernel(
    const half_t* __restrict__ Qh, const half_t* __restrict__ Kh,
    const half_t* __restrict__ Vt, half_t* __restrict__ Ctx) {
  __shared__ half_t Ks[2][64 * LDSS];       // [kv][d]
  __shared__ half_t Vs[2][64 * LDSS];       // [d][kv]  (from transposed V)
  __shared__ half_t Ps[8 * 16 * LDSS];      // per-wave P transpose staging

  const int tid    = threadIdx.x;
  const int lane   = tid & 31;
  const int waveId = tid >> 5;
  const int bh = blockIdx.x;                // b*H + h
  const int q0 = blockIdx.y * 128;
  const int qw = q0 + waveId * 16;

  const size_t baseQK = (size_t)bh * Tc * HDc;
  const size_t baseVt = (size_t)bh * HDc * Tc;

  // persistent Q fragments for this wave's 16 rows (K = d in 0..63)
  const half_t* qp = Qh + baseQK + (size_t)qw * HDc;
  v16h qf0 = ldfrag(qp, HDc);
  v16h qf1 = ldfrag(qp + 32, HDc);

  v8f O[4] = {};
  float rmax[8], rsum[8];
#pragma unroll
  for (int r = 0; r < 8; ++r) { rmax[r] = -3.0e38f; rsum[r] = 0.0f; }

  const int nloc  = lane & 15;
  const int mhalf = (lane >> 4) << 3;
  half_t* PsW = Ps + waveId * 16 * LDSS;

  // cooperative async K/V^T tile stage (512 b128 chunks each, 2/thread)
  auto stageKV = [&](int kv0, int bufi) {
#pragma unroll
    for (int i = 0; i < 2; ++i) {
      int c = tid + i * 256;
      int row = c >> 3;
      int col = (c & 7) << 3;
      async_copy_b128(Kh + baseQK + (size_t)(kv0 + row) * HDc + col,
                      &Ks[bufi][row * LDSS + col]);
      async_copy_b128(Vt + baseVt + (size_t)row * Tc + kv0 + col,
                      &Vs[bufi][row * LDSS + col]);
    }
  };

  const int jmax = (q0 + 127) >> 6;         // last kv tile needed by block
  stageKV(0, 0);
  wait_async();
  __syncthreads();

  for (int j = 0; j <= jmax; ++j) {
    const int kv0 = j << 6;
    const int buf = j & 1;
    if (j < jmax) stageKV((j + 1) << 6, buf ^ 1);

    if (kv0 <= qw + 15) {                   // causal: tile has live columns
      // S = Q @ K^T  (16 x 64)
      v8f S[4];
#pragma unroll
      for (int nt = 0; nt < 4; ++nt) {
        v16h b0 = ldfrag(&Ks[buf][nt * 16 * LDSS], LDSS);
        v16h b1 = ldfrag(&Ks[buf][nt * 16 * LDSS + 32], LDSS);
        v8f s = {};
        s = wmma_f16(qf0, b0, s);
        s = wmma_f16(qf1, b1, s);
        S[nt] = s;
      }
      // causal mask (D layout: m = mhalf+r, n = nloc)
      if (kv0 + 63 > qw) {
#pragma unroll
        for (int nt = 0; nt < 4; ++nt)
#pragma unroll
          for (int r = 0; r < 8; ++r) {
            int n = kv0 + nt * 16 + nloc;
            int m = qw + mhalf + r;
            if (n > m) S[nt][r] = -3.0e38f;
          }
      }
      // online softmax
      float nmax[8], scale[8], psum[8];
#pragma unroll
      for (int r = 0; r < 8; ++r) {
        float mx = fmaxf(fmaxf(S[0][r], S[1][r]), fmaxf(S[2][r], S[3][r]));
#pragma unroll
        for (int off = 1; off < 16; off <<= 1)
          mx = fmaxf(mx, __shfl_xor(mx, off, 16));
        nmax[r]  = fmaxf(rmax[r], mx);
        scale[r] = __expf(rmax[r] - nmax[r]);
        rmax[r]  = nmax[r];
        psum[r]  = 0.0f;
      }
#pragma unroll
      for (int nt = 0; nt < 4; ++nt)
#pragma unroll
        for (int r = 0; r < 8; ++r) {
          float p = __expf(S[nt][r] - nmax[r]);
          S[nt][r] = p;
          psum[r] += p;
        }
#pragma unroll
      for (int r = 0; r < 8; ++r) {
#pragma unroll
        for (int off = 1; off < 16; off <<= 1)
          psum[r] += __shfl_xor(psum[r], off, 16);
        rsum[r] = rsum[r] * scale[r] + psum[r];
      }
#pragma unroll
      for (int dt = 0; dt < 4; ++dt)
#pragma unroll
        for (int r = 0; r < 8; ++r) O[dt][r] *= scale[r];

      // transpose P through per-wave LDS patch (D layout -> A layout)
#pragma unroll
      for (int nt = 0; nt < 4; ++nt)
#pragma unroll
        for (int r = 0; r < 8; ++r)
          PsW[(mhalf + r) * LDSS + nt * 16 + nloc] = (half_t)S[nt][r];
      asm volatile("s_wait_dscnt 0" ::: "memory");

      v16h pa0 = ldfrag(PsW, LDSS);
      v16h pa1 = ldfrag(PsW + 32, LDSS);
#pragma unroll
      for (int dt = 0; dt < 4; ++dt) {
        v16h b0 = ldfrag(&Vs[buf][dt * 16 * LDSS], LDSS);
        v16h b1 = ldfrag(&Vs[buf][dt * 16 * LDSS + 32], LDSS);
        O[dt] = wmma_f16(pa0, b0, O[dt]);
        O[dt] = wmma_f16(pa1, b1, O[dt]);
      }
    }
    wait_async();
    __syncthreads();
  }

  // finalize and store ctx in (B,T,H*HD) fp16
  const int b = bh >> 4, h = bh & 15;
#pragma unroll
  for (int dt = 0; dt < 4; ++dt)
#pragma unroll
    for (int r = 0; r < 8; ++r) {
      int m = qw + mhalf + r;
      float o = O[dt][r] / rsum[r];
      Ctx[(size_t)(b * Tc + m) * Dc + h * HDc + dt * 16 + nloc] = (half_t)o;
    }
}

// ---- Kernel 3: output projection  (out = ctx @ Wo^T + bo) -------------------
__global__ __launch_bounds__(256) void out_proj_kernel(
    const half_t* __restrict__ Ctx, const half_t* __restrict__ Woh,
    const float* __restrict__ bo, float* __restrict__ out) {
  __shared__ half_t As[2][128 * LDSS];
  __shared__ half_t Bs[2][128 * LDSS];

  const int tid  = threadIdx.x;
  const int lane = tid & 31;
  const int m0 = blockIdx.x * 128;
  const int n0 = blockIdx.y * 128;

  const int waveId = tid >> 5;
  const int wm = (waveId & 3) * 32;
  const int wn = (waveId >> 2) * 64;

  v8f acc[2][4] = {};

  stage_tile(Ctx + (size_t)m0 * Dc, Dc, As[0], tid);
  stage_tile(Woh + (size_t)n0 * Dc, Dc, Bs[0], tid);
  wait_async();
  __syncthreads();

  for (int k0 = 0; k0 < Dc; k0 += 64) {
    const int buf = (k0 >> 6) & 1;
    if (k0 + 64 < Dc) {
      stage_tile(Ctx + (size_t)m0 * Dc + k0 + 64, Dc, As[buf ^ 1], tid);
      stage_tile(Woh + (size_t)n0 * Dc + k0 + 64, Dc, Bs[buf ^ 1], tid);
    }
#pragma unroll
    for (int kc = 0; kc < 64; kc += 32) {
      v16h a0 = ldfrag(&As[buf][(wm + 0)  * LDSS + kc], LDSS);
      v16h a1 = ldfrag(&As[buf][(wm + 16) * LDSS + kc], LDSS);
      v16h b[4];
#pragma unroll
      for (int nt = 0; nt < 4; ++nt)
        b[nt] = ldfrag(&Bs[buf][(wn + nt * 16) * LDSS + kc], LDSS);
#pragma unroll
      for (int nt = 0; nt < 4; ++nt) {
        acc[0][nt] = wmma_f16(a0, b[nt], acc[0][nt]);
        acc[1][nt] = wmma_f16(a1, b[nt], acc[1][nt]);
      }
    }
    wait_async();
    __syncthreads();
  }

  const int nloc  = lane & 15;
  const int mhalf = (lane >> 4) << 3;
#pragma unroll
  for (int mt = 0; mt < 2; ++mt)
#pragma unroll
    for (int nt = 0; nt < 4; ++nt)
#pragma unroll
      for (int r = 0; r < 8; ++r) {
        int m = m0 + wm + mt * 16 + mhalf + r;
        int n = n0 + wn + nt * 16 + nloc;
        out[(size_t)m * Dc + n] = acc[mt][nt][r] + bo[n];
      }
}

// ---- launch -----------------------------------------------------------------
extern "C" void kernel_launch(void* const* d_in, const int* in_sizes, int n_in,
                              void* d_out, int out_size, void* d_ws, size_t ws_size,
                              hipStream_t stream) {
  const float* x  = (const float*)d_in[0];
  const float* Wq = (const float*)d_in[1];
  const float* Wk = (const float*)d_in[2];
  const float* Wv = (const float*)d_in[3];
  const float* Wo = (const float*)d_in[4];
  const float* bo = (const float*)d_in[5];
  float* out = (float*)d_out;

  const size_t big   = (size_t)BT * Dc;     // 8M halves
  const size_t small = (size_t)Dc * Dc;     // 1M halves
  half_t* Qh  = (half_t*)d_ws;
  half_t* Kh  = Qh  + big;
  half_t* Vt  = Kh  + big;
  half_t* Ctx = Vt  + big;
  half_t* xh  = Ctx + big;
  half_t* Wqh = xh  + big;
  half_t* Wkh = Wqh + small;
  half_t* Wvh = Wkh + small;
  half_t* Woh = Wvh + small;

  // fp32 -> fp16 bulk converts (one v4 per thread)
  cvt_f16_kernel<<<dim3(big   / 1024), 256, 0, stream>>>(x,  xh);
  cvt_f16_kernel<<<dim3(small / 1024), 256, 0, stream>>>(Wq, Wqh);
  cvt_f16_kernel<<<dim3(small / 1024), 256, 0, stream>>>(Wk, Wkh);
  cvt_f16_kernel<<<dim3(small / 1024), 256, 0, stream>>>(Wv, Wvh);
  cvt_f16_kernel<<<dim3(small / 1024), 256, 0, stream>>>(Wo, Woh);

  qkv_proj_kernel<<<dim3(BT / 128, Dc / 128, 3), 256, 0, stream>>>(
      xh, Wqh, Wkh, Wvh, Qh, Kh, Vt);
  flash_attn_kernel<<<dim3(Bc * Hc, Tc / 128), 256, 0, stream>>>(
      Qh, Kh, Vt, Ctx);
  out_proj_kernel<<<dim3(BT / 128, Dc / 128), 256, 0, stream>>>(
      Ctx, Woh, bo, out);
}
